// BLSTM5_28269474742714
// MI455X (gfx1250) — matrix operand order repaired and checked by reading, not compiled, not run on hardware
//
#include <hip/hip_runtime.h>
#include <hip/hip_bf16.h>
#include <math.h>

typedef __bf16 bf16;
typedef __attribute__((ext_vector_type(16))) __bf16 v16bf;
typedef __attribute__((ext_vector_type(8)))  float  v8f;
typedef __attribute__((ext_vector_type(4)))  float  vf4;

#define T_STEPS 300
#define BATCH   64
#define HID     512
#define NGATE   2048   // 4*HID
#define FEAT    128
#define LABEL   1251

union Frag { v16bf v; vf4 f[2]; };
struct __align__(16) bh8 { bf16 v[8]; };   // 8 packed bf16 = one b128

__device__ __forceinline__ float sigm(float x) { return 1.0f / (1.0f + __expf(-x)); }

// ---- CDNA5 async global<->LDS ops (ASYNCcnt path), per CDNA5 ISA ----
__device__ __forceinline__ void async_ld16(const bf16* lds_dst, const bf16* gsrc) {
    asm volatile("global_load_async_to_lds_b128 %0, %1, off"
                 :: "v"((unsigned)(unsigned long long)(uintptr_t)lds_dst),
                    "v"(gsrc)
                 : "memory");
}
__device__ __forceinline__ void async_st16(const bf16* gdst, const bf16* lds_src) {
    asm volatile("global_store_async_from_lds_b128 %0, %1, off"
                 :: "v"(gdst),
                    "v"((unsigned)(unsigned long long)(uintptr_t)lds_src)
                 : "memory");
}
__device__ __forceinline__ void wait_async0() {
    asm volatile("s_wait_asynccnt 0" ::: "memory");
}
__device__ __forceinline__ void wait_async4() {
    asm volatile("s_wait_asynccnt 4" ::: "memory");
}

// A fragment (16x32 bf16) from a row-major tile, base = tile origin (row stride ld elems)
// lanes 0-15: M=lane, K {0..7,16..23}; lanes 16-31: M=lane-16, K {8..15,24..31}
__device__ __forceinline__ v16bf load_a_frag(const bf16* base, int ld, int lane) {
    int row = lane & 15;
    int kh  = (lane >> 4) * 8;
    const bf16* p = base + row * ld + kh;
    Frag fr;
    fr.f[0] = *(const vf4*)(p);
    fr.f[1] = *(const vf4*)(p + 16);
    return fr.v;
}

// B fragment (32x16 bf16) from N-major (pre-transposed) weights: base = WT + n0*ldk + k0
// lanes 0-15: N=lane, K {0..7,16..23}; lanes 16-31: N=lane-16, K {8..15,24..31}
__device__ __forceinline__ v16bf load_b_frag(const bf16* base, int ldk, int lane) {
    int col = lane & 15;
    int kh  = (lane >> 4) * 8;
    const bf16* p = base + (size_t)col * ldk + kh;
    Frag fr;
    fr.f[0] = *(const vf4*)(p);
    fr.f[1] = *(const vf4*)(p + 16);
    return fr.v;
}

// ---- X1 [B,F,T,1] f32  ->  x0 [T,B,F] bf16 ----
__global__ void convx_kernel(const float* __restrict__ X1, bf16* __restrict__ x0) {
    int idx = blockIdx.x * 256 + threadIdx.x;
    if (idx >= BATCH * FEAT * T_STEPS) return;
    int t = idx % T_STEPS;
    int f = (idx / T_STEPS) % FEAT;
    int b = idx / (T_STEPS * FEAT);
    x0[((size_t)t * BATCH + b) * FEAT + f] = (bf16)X1[idx];
}

// ---- W f32 [Din+512, 2048]  ->  WxT bf16 [2048, Din], WhT bf16 [2048, 512] ----
__global__ void wconv_kernel(const float* __restrict__ Wf, const float* __restrict__ Wb, int Din,
                             bf16* __restrict__ WxTf, bf16* __restrict__ WxTb,
                             bf16* __restrict__ WhTf, bf16* __restrict__ WhTb) {
    const float* W = blockIdx.z ? Wb : Wf;
    bf16* WxT = blockIdx.z ? WxTb : WxTf;
    bf16* WhT = blockIdx.z ? WhTb : WhTf;
    int n = blockIdx.x;          // 0..2047
    int K = Din + HID;
    for (int k = threadIdx.x; k < K; k += blockDim.x) {
        bf16 v = (bf16)W[(size_t)k * NGATE + n];
        if (k < Din) WxT[(size_t)n * Din + k] = v;
        else         WhT[(size_t)n * HID + (k - Din)] = v;
    }
}

// ---- Time-parallel input GEMM: xpart[t][n][b] = (in[t,b,:] @ Wx + bias), bf16, batch-minor ----
// grid: (2048/64, 19200/128, 2 dirs), block 256 (8 waves, each 2x2 16x16 tiles)
__global__ __launch_bounds__(256) void xpart_gemm(
    const bf16* __restrict__ in, int Din,
    const bf16* __restrict__ WxTf, const bf16* __restrict__ WxTb,
    const float* __restrict__ biasf, const float* __restrict__ biasb,
    bf16* __restrict__ xpf, bf16* __restrict__ xpb)
{
    __shared__ __align__(16) bf16 Abuf[2][128 * 32];
    const int dir = blockIdx.z;
    const bf16*  WxT  = dir ? WxTb  : WxTf;
    const float* bias = dir ? biasb : biasf;
    bf16*        xp   = dir ? xpb   : xpf;

    const int n_base = blockIdx.x * 64;
    const int m_base = blockIdx.y * 128;
    const int tid  = threadIdx.x;
    const int w    = tid >> 5;
    const int lane = tid & 31;
    const int wm = (w & 3) * 32;   // wave row offset within 128
    const int wn = (w >> 2) * 32;  // wave col offset within 64

    v8f acc[2][2];
    v8f zero8 = {0.f, 0.f, 0.f, 0.f, 0.f, 0.f, 0.f, 0.f};
    acc[0][0] = zero8; acc[0][1] = zero8; acc[1][0] = zero8; acc[1][1] = zero8;

    const int nk = Din >> 5;
    {   // stage chunk 0 (512 16B units across 256 threads, 2 each)
        int r0 = tid >> 2, q0 = tid & 3;
        int r1 = (tid + 256) >> 2, q1 = (tid + 256) & 3;
        async_ld16(&Abuf[0][r0 * 32 + q0 * 8], in + (size_t)(m_base + r0) * Din + q0 * 8);
        async_ld16(&Abuf[0][r1 * 32 + q1 * 8], in + (size_t)(m_base + r1) * Din + q1 * 8);
    }
    for (int kk = 0; kk < nk; ++kk) {
        wait_async0();
        __syncthreads();
        if (kk + 1 < nk) {
            int k0n  = (kk + 1) << 5;
            int bufn = (kk + 1) & 1;
            int r0 = tid >> 2, q0 = tid & 3;
            int r1 = (tid + 256) >> 2, q1 = (tid + 256) & 3;
            async_ld16(&Abuf[bufn][r0 * 32 + q0 * 8],
                       in + (size_t)(m_base + r0) * Din + k0n + q0 * 8);
            async_ld16(&Abuf[bufn][r1 * 32 + q1 * 8],
                       in + (size_t)(m_base + r1) * Din + k0n + q1 * 8);
        }
        const int buf = kk & 1;
        const int k0  = kk << 5;
        v16bf afrag[2], bfrag[2];
        afrag[0] = load_a_frag(&Abuf[buf][(wm + 0)  * 32], 32, lane);
        afrag[1] = load_a_frag(&Abuf[buf][(wm + 16) * 32], 32, lane);
        bfrag[0] = load_b_frag(WxT + (size_t)(n_base + wn + 0)  * Din + k0, Din, lane);
        bfrag[1] = load_b_frag(WxT + (size_t)(n_base + wn + 16) * Din + k0, Din, lane);
#pragma unroll
        for (int mi = 0; mi < 2; ++mi)
#pragma unroll
            for (int ni = 0; ni < 2; ++ni)
                acc[mi][ni] = __builtin_amdgcn_wmma_f32_16x16x32_bf16(
                    false, afrag[mi], false, bfrag[ni], (short)0, acc[mi][ni], false, false);
    }
    // epilogue: +bias, pack 8 consecutive batch rows per lane -> one b128 store (batch-minor)
#pragma unroll
    for (int mi = 0; mi < 2; ++mi) {
#pragma unroll
        for (int ni = 0; ni < 2; ++ni) {
            int ncol  = n_base + wn + ni * 16 + (lane & 15);
            int mrow0 = m_base + wm + mi * 16 + ((lane >> 4) * 8);  // 8 consecutive rows
            int tIdx  = mrow0 >> 6;
            int b0    = mrow0 & 63;
            float bv  = bias[ncol];
            bh8 o;
#pragma unroll
            for (int r = 0; r < 8; ++r) o.v[r] = (bf16)(acc[mi][ni][r] + bv);
            *(bh8*)(xp + ((size_t)tIdx * NGATE + ncol) * 64 + b0) = o;
        }
    }
}

// ---- Persistent recurrent scan: one 32-wave block per direction ----
// Wave (m,g) owns a 16-batch x 64-hidden block of all 4 gates. c in registers;
// h ping-pongs in LDS; gate K-loop split into two passes (i,j then f,o) to
// halve live accumulators; h(t) -> outseq via async LDS->global stores.
__global__ __launch_bounds__(1024) void lstm_scan(
    const bf16* __restrict__ xpf, const bf16* __restrict__ xpb,
    const bf16* __restrict__ WhTf, const bf16* __restrict__ WhTb,
    bf16* __restrict__ outseq)
{
    extern __shared__ __align__(16) bf16 hbuf[];   // [2][64*512]
    const int dir = blockIdx.x;
    const bf16* xp  = dir ? xpb  : xpf;
    const bf16* WhT = dir ? WhTb : WhTf;
    const int tid  = threadIdx.x;
    const int w    = tid >> 5;
    const int lane = tid & 31;
    const int m = (w & 3);    // 16-row batch block
    const int g = (w >> 2);   // 64-col hidden group

    for (int i = tid; i < 2 * BATCH * HID; i += 1024) hbuf[i] = (bf16)0.0f;

    float cst[4][8];
#pragma unroll
    for (int j = 0; j < 4; ++j)
#pragma unroll
        for (int r = 0; r < 8; ++r) cst[j][r] = 0.0f;

    __syncthreads();

    const int colbase = g * 64 + (lane & 15);
    const int rowb0   = m * 16 + ((lane >> 4) * 8);   // 8 consecutive batch rows per lane
    v8f zero8 = {0.f, 0.f, 0.f, 0.f, 0.f, 0.f, 0.f, 0.f};

    for (int step = 0; step < T_STEPS; ++step) {
        const int t    = dir ? (T_STEPS - 1 - step) : step;
        const int cur  = step & 1;
        const int prev = cur ^ 1;
        const bf16* hprev = &hbuf[prev * (BATCH * HID) + (m * 16) * HID];
        const size_t tb = (size_t)t * NGATE;

        // ---- pass 1: gates i (0) and j (1) ----
        v8f acc[2][4];
#pragma unroll
        for (int gi = 0; gi < 2; ++gi)
#pragma unroll
            for (int j = 0; j < 4; ++j) acc[gi][j] = zero8;
        for (int k0 = 0; k0 < HID; k0 += 32) {
            v16bf afrag = load_a_frag(hprev + k0, HID, lane);
#pragma unroll
            for (int gi = 0; gi < 2; ++gi)
#pragma unroll
                for (int j = 0; j < 4; ++j) {
                    int n0 = gi * HID + g * 64 + j * 16;
                    v16bf bfrag = load_b_frag(WhT + (size_t)n0 * HID + k0, HID, lane);
                    acc[gi][j] = __builtin_amdgcn_wmma_f32_16x16x32_bf16(
                        false, afrag, false, bfrag, (short)0, acc[gi][j], false, false);
                }
        }
        float u[4][8];
#pragma unroll
        for (int j = 0; j < 4; ++j) {
            const bh8 xi = *(const bh8*)(xp + (tb + 0 * HID + colbase + j * 16) * 64 + rowb0);
            const bh8 xj = *(const bh8*)(xp + (tb + 1 * HID + colbase + j * 16) * 64 + rowb0);
#pragma unroll
            for (int r = 0; r < 8; ++r) {
                float ig = acc[0][j][r] + (float)xi.v[r];
                float jg = acc[1][j][r] + (float)xj.v[r];
                u[j][r] = sigm(ig) * tanhf(jg);
            }
        }

        // ---- pass 2: gates f (2) and o (3) ----
#pragma unroll
        for (int gi = 0; gi < 2; ++gi)
#pragma unroll
            for (int j = 0; j < 4; ++j) acc[gi][j] = zero8;
        for (int k0 = 0; k0 < HID; k0 += 32) {
            v16bf afrag = load_a_frag(hprev + k0, HID, lane);
#pragma unroll
            for (int gi = 0; gi < 2; ++gi)
#pragma unroll
                for (int j = 0; j < 4; ++j) {
                    int n0 = (2 + gi) * HID + g * 64 + j * 16;
                    v16bf bfrag = load_b_frag(WhT + (size_t)n0 * HID + k0, HID, lane);
                    acc[gi][j] = __builtin_amdgcn_wmma_f32_16x16x32_bf16(
                        false, afrag, false, bfrag, (short)0, acc[gi][j], false, false);
                }
        }

        // ---- cell update + h writeback to LDS ----
#pragma unroll
        for (int j = 0; j < 4; ++j) {
            const bh8 xf = *(const bh8*)(xp + (tb + 2 * HID + colbase + j * 16) * 64 + rowb0);
            const bh8 xo = *(const bh8*)(xp + (tb + 3 * HID + colbase + j * 16) * 64 + rowb0);
#pragma unroll
            for (int r = 0; r < 8; ++r) {
                float fg = acc[0][j][r] + (float)xf.v[r];
                float og = acc[1][j][r] + (float)xo.v[r];
                float c = sigm(fg + 1.0f) * cst[j][r] + u[j][r];
                float h = sigm(og) * tanhf(c);
                cst[j][r] = c;
                hbuf[cur * (BATCH * HID) + (rowb0 + r) * HID + colbase + j * 16] = (bf16)h;
            }
        }

        // stores issued 2 steps ago target the buffer we'll overwrite next step:
        // force them complete (own wave) before the barrier makes it visible to all.
        wait_async4();
        __syncthreads();

        // ---- h(t) -> outseq[t] via async LDS->global b128 stores (4 per thread) ----
        const size_t trow = (size_t)t * BATCH;
        const bf16* srcbase = &hbuf[cur * (BATCH * HID)];
#pragma unroll
        for (int pass = 0; pass < 4; ++pass) {
            int chunk = tid + pass * 1024;        // 0..4095 chunks of 8 elems
            int bb = chunk >> 6;
            int nc = (chunk & 63) * 8;
            async_st16(outseq + (trow + bb) * (2 * HID) + dir * HID + nc,
                       srcbase + bb * HID + nc);
        }
    }
}

// ---- last timestep -> L2 normalize -> [64,1024] @ W1 + b1 ----
__global__ __launch_bounds__(256) void final_kernel(
    const bf16* __restrict__ seq, const float* __restrict__ W1,
    const float* __restrict__ b1, float* __restrict__ out)
{
    __shared__ float row[2 * HID];
    __shared__ float red[256];
    int b = blockIdx.x;
    const bf16* src = seq + ((size_t)(T_STEPS - 1) * BATCH + b) * (2 * HID);
    float ss = 0.f;
    for (int k = threadIdx.x; k < 2 * HID; k += 256) {
        float v = (float)src[k];
        row[k] = v;
        ss += v * v;
    }
    red[threadIdx.x] = ss;
    __syncthreads();
    for (int s = 128; s > 0; s >>= 1) {
        if (threadIdx.x < s) red[threadIdx.x] += red[threadIdx.x + s];
        __syncthreads();
    }
    float scale = rsqrtf(fmaxf(red[0], 1e-12f));
    for (int l = threadIdx.x; l < LABEL; l += 256) {
        float acc = b1[l];
        for (int k = 0; k < 2 * HID; ++k)
            acc += row[k] * scale * W1[(size_t)k * LABEL + l];
        out[(size_t)b * LABEL + l] = acc;
    }
}

extern "C" void kernel_launch(void* const* d_in, const int* in_sizes, int n_in,
                              void* d_out, int out_size, void* d_ws, size_t ws_size,
                              hipStream_t stream)
{
    (void)in_sizes; (void)n_in; (void)out_size; (void)ws_size;
    const float* X1    = (const float*)d_in[0];
    const float* W_fw0 = (const float*)d_in[1];
    const float* b_fw0 = (const float*)d_in[2];
    const float* W_bw0 = (const float*)d_in[3];
    const float* b_bw0 = (const float*)d_in[4];
    const float* W_fwr = (const float*)d_in[5];
    const float* b_fwr = (const float*)d_in[6];
    const float* W_bwr = (const float*)d_in[7];
    const float* b_bwr = (const float*)d_in[8];
    const float* W1    = (const float*)d_in[9];
    const float* b1    = (const float*)d_in[10];
    float* out = (float*)d_out;

    char* ws = (char*)d_ws;
    size_t off = 0;
    auto alloc = [&](size_t bytes) -> char* {
        char* p = ws + off;
        off += (bytes + 255) & ~(size_t)255;
        return p;
    };
    bf16* x0   = (bf16*)alloc((size_t)T_STEPS * BATCH * FEAT  * 2);
    bf16* seqA = (bf16*)alloc((size_t)T_STEPS * BATCH * 1024  * 2);
    bf16* seqB = (bf16*)alloc((size_t)T_STEPS * BATCH * 1024  * 2);
    bf16* xpf  = (bf16*)alloc((size_t)T_STEPS * BATCH * NGATE * 2);
    bf16* xpb  = (bf16*)alloc((size_t)T_STEPS * BATCH * NGATE * 2);
    bf16* WxTf = (bf16*)alloc((size_t)NGATE * 1024 * 2);
    bf16* WxTb = (bf16*)alloc((size_t)NGATE * 1024 * 2);
    bf16* WhTf = (bf16*)alloc((size_t)NGATE * HID * 2);
    bf16* WhTb = (bf16*)alloc((size_t)NGATE * HID * 2);

    {
        int total = BATCH * FEAT * T_STEPS;
        convx_kernel<<<(total + 255) / 256, 256, 0, stream>>>(X1, x0);
    }

    struct LayerDesc {
        const float *Wf, *Wb, *bfp, *bbp;
        int Din;
        const bf16* in;
        bf16* outp;
    };
    const size_t wrstride = (size_t)(3 * HID) * NGATE;  // [1536,2048] per layer
    LayerDesc L[3] = {
        { W_fw0,              W_bw0,              b_fw0,           b_bw0,           FEAT, x0,   seqA },
        { W_fwr + 0*wrstride, W_bwr + 0*wrstride, b_fwr + 0*NGATE, b_bwr + 0*NGATE, 1024, seqA, seqB },
        { W_fwr + 1*wrstride, W_bwr + 1*wrstride, b_fwr + 1*NGATE, b_bwr + 1*NGATE, 1024, seqB, seqA },
    };

    for (int l = 0; l < 3; ++l) {
        wconv_kernel<<<dim3(NGATE, 1, 2), 256, 0, stream>>>(
            L[l].Wf, L[l].Wb, L[l].Din, WxTf, WxTb, WhTf, WhTb);
        xpart_gemm<<<dim3(NGATE / 64, (T_STEPS * BATCH) / 128, 2), 256, 0, stream>>>(
            L[l].in, L[l].Din, WxTf, WxTb, L[l].bfp, L[l].bbp, xpf, xpb);
        lstm_scan<<<dim3(2), dim3(1024), 2 * BATCH * HID * (int)sizeof(bf16), stream>>>(
            xpf, xpb, WhTf, WhTb, L[l].outp);
    }

    final_kernel<<<dim3(BATCH), 256, 0, stream>>>(seqA, W1, b1, out);
}
